// NCA_51273319579666
// MI455X (gfx1250) — matrix operand order repaired and checked by reading, not compiled, not run on hardware
//
#include <hip/hip_runtime.h>
#include <hip/hip_bf16.h>

typedef __attribute__((ext_vector_type(16))) __bf16 v16bf;
typedef __attribute__((ext_vector_type(8)))  float  v8f;

// Problem constants (from reference): B=16, C=16, H=W=256, HID=128, 3C=48 -> pad K to 64.
#define HDIM 256
#define WDIM 256

// ---- CDNA5 WMMA fragment index maps (wave32, 16-bit elements) ----
// A matrix 16x32 (MxK): lane 0-15 = M, VGPR0-3 hold K{0..7}(+8 for hi half),
// VGPR4-7 hold K{16..23}(+8 for hi half).  e = element 0..15 within v16bf.
__device__ __forceinline__ int kmapA(int e, int lane) {
  int v = e >> 1, p = e & 1, half = lane >> 4;
  return ((v & 4) << 2) | (half << 3) | ((v & 3) << 1) | p;
}
// B matrix 32x16 (KxN): N = lane&15; same K order as A:
// VGPR0-3 hold K{0..7}(+8 for hi half), VGPR4-7 hold K{16..23}(+8 for hi half).
__device__ __forceinline__ int kmapB(int e, int lane) {
  int v = e >> 1, p = e & 1, half = lane >> 4;
  return ((v & 4) << 2) | (half << 3) | ((v & 3) << 1) | p;
}

__device__ __forceinline__ float ld_x(const float* __restrict__ x, int b, int c,
                                      int h, int w) {
  if ((unsigned)h >= (unsigned)HDIM || (unsigned)w >= (unsigned)WDIM) return 0.f;
  return x[(((b << 4) + c) << 16) + (h << 8) + w];
}

// Perception: f=0 identity, f=1 sobel-x, f=2 sobel-y (correlation, zero pad, /8)
__device__ __forceinline__ float percept(const float* __restrict__ x, int b, int c,
                                         int f, int h, int w) {
  if (f == 0) return ld_x(x, b, c, h, w);
  if (f == 1)
    return (-ld_x(x,b,c,h-1,w-1) +      ld_x(x,b,c,h-1,w+1)
            -2.f*ld_x(x,b,c,h  ,w-1) + 2.f*ld_x(x,b,c,h  ,w+1)
            -ld_x(x,b,c,h+1,w-1) +      ld_x(x,b,c,h+1,w+1)) * 0.125f;
  return (-ld_x(x,b,c,h-1,w-1) - 2.f*ld_x(x,b,c,h-1,w) - ld_x(x,b,c,h-1,w+1)
          + ld_x(x,b,c,h+1,w-1) + 2.f*ld_x(x,b,c,h+1,w) + ld_x(x,b,c,h+1,w+1)) * 0.125f;
}

// LDS layout (bytes):
//   [0,16384)   yfrag  : A-fragments of perception, [8 groups][2 ksteps][32 lanes][16] bf16
//   [16384,32768) w0frag: B-fragments of w0^T,      [8 ntiles][2 ksteps][32][16] bf16
//   [0,32768)   hfrag  : (phase 2, overlaps above)  [8 waves][4 ksteps][32][16] bf16
//   [32768,36864) w1frag: B-fragments of w1^T,      [4 ksteps][32][16] bf16
//   [36864,45056) dxbuf : [8 waves][16 pix][16 ch] f32
__global__ __launch_bounds__(256) void nca_step_kernel(
    const float* __restrict__ xg,  const float* __restrict__ w0g,
    const float* __restrict__ b0g, const float* __restrict__ w1g,
    const float* __restrict__ b1g, const float* __restrict__ fmg,
    float* __restrict__ outg, float* __restrict__ alphag) {
  __shared__ alignas(32) unsigned char smem[45056];
  __bf16* yfrag  = (__bf16*)(smem);
  __bf16* w0frag = (__bf16*)(smem + 16384);
  __bf16* hfrag  = (__bf16*)(smem);          // phase-2 reuse of yfrag+w0frag
  __bf16* w1frag = (__bf16*)(smem + 32768);
  float*  dxbuf  = (float*)(smem + 36864);

  const int tid  = threadIdx.x;
  const int lane = tid & 31;
  const int wv   = tid >> 5;        // wave 0..7 -> pixel group of 16
  const int half = lane >> 4;
  const int n16  = lane & 15;

  const int blk   = blockIdx.x;     // 2 * 256 * 16 = 8192 blocks
  const int wbase = (blk & 1) << 7; // 128-pixel tile in W
  const int h     = (blk >> 1) & 255;
  const int b     = blk >> 9;

  // ---- fill B fragments for GEMM1: B[k][n] = w0[n][k], K padded 48->64 ----
  for (int i = tid; i < 8192; i += 256) {
    int e = i & 15, l = (i >> 4) & 31, s = (i >> 9) & 1, nt = i >> 10;
    int k = kmapB(e, l) + (s << 5);
    int n = (nt << 4) + (l & 15);
    w0frag[i] = (k < 48) ? (__bf16)w0g[n * 48 + k] : (__bf16)0.f;
  }
  // ---- fill B fragments for GEMM2: B[k][n] = w1[n][k], K=128 ----
  for (int i = tid; i < 2048; i += 256) {
    int e = i & 15, l = (i >> 4) & 31, s = i >> 9;
    int k = kmapB(e, l) + (s << 5);
    w1frag[i] = (__bf16)w1g[(l & 15) * 128 + k];
  }
  // ---- perception directly into A-fragment slots (L2-resident x reads) ----
  for (int i = tid; i < 8192; i += 256) {
    int e = i & 15, l = (i >> 4) & 31, s = (i >> 9) & 1, g = i >> 10;
    int k = kmapA(e, l) + (s << 5);
    float v = 0.f;
    if (k < 48) {
      int c = k / 3, f = k - 3 * c;            // y channel j = 3c + f
      int w = wbase + (g << 4) + (l & 15);
      v = percept(xg, b, c, f, h, w);
    }
    yfrag[i] = (__bf16)v;
  }
  __syncthreads();

  // ---- GEMM1: h[16x128] = y[16x64] * w0^T[64x128] ----
  v16bf a0 = *(const v16bf*)(yfrag + ((((wv << 1) + 0) << 5) + lane) * 16);
  v16bf a1 = *(const v16bf*)(yfrag + ((((wv << 1) + 1) << 5) + lane) * 16);
  v8f acc[8];
#pragma unroll
  for (int nt = 0; nt < 8; ++nt) {
    v8f c = {};
    v16bf bA = *(const v16bf*)(w0frag + ((((nt << 1) + 0) << 5) + lane) * 16);
    c = __builtin_amdgcn_wmma_f32_16x16x32_bf16(false, a0, false, bA,
                                                (short)0, c, false, false);
    v16bf bB = *(const v16bf*)(w0frag + ((((nt << 1) + 1) << 5) + lane) * 16);
    c = __builtin_amdgcn_wmma_f32_16x16x32_bf16(false, a1, false, bB,
                                                (short)0, c, false, false);
    acc[nt] = c;
  }
  __syncthreads();  // all waves done reading yfrag/w0frag -> reuse as hfrag

  // ---- bias + ReLU, scatter h (D layout) into GEMM2 A-fragment layout ----
#pragma unroll
  for (int nt = 0; nt < 8; ++nt) {
#pragma unroll
    for (int r = 0; r < 8; ++r) {
      int m  = r + (half << 3);       // pixel row within wave tile
      int kg = (nt << 4) + n16;       // hidden channel 0..127
      float hv = fmaxf(acc[nt][r] + b0g[kg], 0.f);
      int kk = kg & 31, ks = kg >> 5;
      int h2 = (kk >> 3) & 1;                        // inverse of kmapA
      int v2 = ((kk & 7) >> 1) + ((kk >> 4) << 2);
      int e2 = (v2 << 1) + (kk & 1);
      int l2 = m + (h2 << 4);
      hfrag[((((wv << 2) + ks) << 5) + l2) * 16 + e2] = (__bf16)hv;
    }
  }

  // ---- GEMM2: dx[16x16] = h[16x128] * w1^T[128x16] (wave-local, DS in-order) ----
  v8f c2 = {};
#pragma unroll
  for (int s = 0; s < 4; ++s) {
    v16bf ah = *(const v16bf*)(hfrag + ((((wv << 2) + s) << 5) + lane) * 16);
    v16bf bh = *(const v16bf*)(w1frag + ((s << 5) + lane) * 16);
    c2 = __builtin_amdgcn_wmma_f32_16x16x32_bf16(false, ah, false, bh,
                                                 (short)0, c2, false, false);
  }

  // ---- dx -> LDS transpose for coalesced channel-plane stores ----
#pragma unroll
  for (int r = 0; r < 8; ++r) {
    int m = r + (half << 3);
    dxbuf[(((wv << 4) + m) << 4) + n16] = c2[r] + b1g[n16];
  }

  // ---- x_new = x + fire_gate * dx ; also emit new alpha plane ----
  {
    int p = n16;                       // pixel within wave tile
    int w = wbase + (wv << 4) + p;
    float fm = fmg[(b << 16) + (h << 8) + w];
    float gate = (fm <= 0.5f) ? 1.f : 0.f;
#pragma unroll
    for (int i = 0; i < 8; ++i) {
      int c = (i << 1) + half;         // half-waves cover even/odd channels
      float dv = dxbuf[(((wv << 4) + p) << 4) + c];
      int gi = (((b << 4) + c) << 16) + (h << 8) + w;
      float nv = xg[gi] + gate * dv;
      outg[gi] = nv;
      if (c == 3) alphag[(b << 16) + (h << 8) + w] = nv;
    }
  }
}

// Kernel 2: alive mask = (maxpool3x3(alpha_old) > 0.1) | (maxpool3x3(alpha_new) > 0.1).
// Zero out dead pixels in-place (each thread touches only its own pixel's channels).
__global__ __launch_bounds__(256) void nca_mask_kernel(
    const float* __restrict__ xg, const float* __restrict__ alphag,
    float* __restrict__ outg) {
  int pix = blockIdx.x * 256 + threadIdx.x;   // 16*256*256 pixels
  int b = pix >> 16;
  int hw = pix & 65535;
  int h = hw >> 8, w = hw & 255;
  const float* xa = xg + (((b << 4) + 3) << 16);
  const float* na = alphag + (b << 16);
  float mpre = -3.0e38f, mpost = -3.0e38f;
#pragma unroll
  for (int dh = -1; dh <= 1; ++dh) {
    int hh = h + dh;
    if ((unsigned)hh >= 256u) continue;
#pragma unroll
    for (int dw = -1; dw <= 1; ++dw) {
      int ww = w + dw;
      if ((unsigned)ww >= 256u) continue;
      int o = (hh << 8) + ww;
      mpre  = fmaxf(mpre,  xa[o]);
      mpost = fmaxf(mpost, na[o]);
    }
  }
  if (!(mpre > 0.1f || mpost > 0.1f)) {
#pragma unroll
    for (int c = 0; c < 16; ++c)
      outg[(((b << 4) + c) << 16) + hw] = 0.f;
  }
}

extern "C" void kernel_launch(void* const* d_in, const int* in_sizes, int n_in,
                              void* d_out, int out_size, void* d_ws, size_t ws_size,
                              hipStream_t stream) {
  const float* x  = (const float*)d_in[0];
  const float* w0 = (const float*)d_in[1];
  const float* b0 = (const float*)d_in[2];
  const float* w1 = (const float*)d_in[3];
  const float* b1 = (const float*)d_in[4];
  const float* fm = (const float*)d_in[5];
  float* out = (float*)d_out;
  float* alpha_ws = (float*)d_ws;   // 16*256*256 floats = 4 MiB scratch

  nca_step_kernel<<<8192, 256, 0, stream>>>(x, w0, b0, w1, b1, fm, out, alpha_ws);
  nca_mask_kernel<<<4096, 256, 0, stream>>>(x, alpha_ws, out);
}
